// MutualInformation_1116691497092
// MI455X (gfx1250) — matrix operand order repaired and checked by reading, not compiled, not run on hardware
//
#include <hip/hip_runtime.h>
#include <math.h>

// ---------------- constants ----------------
#define BINS      64
#define SIGMA_F   3.0f
#define EPS_F     1e-10f
#define CHUNK     128           // samples per staged chunk (4 WMMA K-steps of 32)
#define TWO_PI_F  6.28318530717958647692f

// workspace layout (float offsets into d_ws)
#define WS_SCALARS 0      // 8 floats: mnFm,mxFm,mnWm,mxWm,mnFa,mxFa,mnWa,mxWa
#define WS_WSUM    8      // 1 float
#define WS_ANY     9      // 1 int  (any masked voxel?)
#define WS_BINSF   16     // 64
#define WS_BINSW   80     // 64
#define WS_MARGF   144    // 64  (sum of weighted p_f over samples)
#define WS_MARGW   208    // 64  (sum of weighted p_w over samples)
#define WS_JOINT   272    // 64*64 = 4096
#define WS_PART    4368   // 1024 blocks * 8 floats = 8192
#define WS_PANY    12560  // 1024 ints
#define WS_SAMPLES 16384  // interleaved samples: {xs_f, xs_w, w} * m  (3m floats)

typedef __attribute__((ext_vector_type(16))) _Float16 v16h;
typedef __attribute__((ext_vector_type(8)))  _Float16 v8h;
typedef __attribute__((ext_vector_type(8)))  float    v8f;
typedef __attribute__((ext_vector_type(4)))  unsigned u32x4;
typedef __attribute__((ext_vector_type(8)))  int      i32x8;
typedef __attribute__((ext_vector_type(4)))  int      i32x4;

// Tensor Data Mover available? (device pass on toolchains that declare it)
#if defined(__AMDGCN__) && __has_builtin(__builtin_amdgcn_tensor_load_to_lds)
#define USE_TDM 1
#else
#define USE_TDM 0
#endif

__device__ __forceinline__ unsigned hash32(unsigned x) {
    x ^= x >> 16; x *= 0x7feb352du;
    x ^= x >> 15; x *= 0x846ca68bu;
    x ^= x >> 16; return x;
}

__device__ __forceinline__ float blockReduceOp(float v, int op /*0=min,1=max,2=sum*/,
                                               float* red, int nthreads) {
    int t = threadIdx.x;
    red[t] = v;
    __syncthreads();
    for (int s = nthreads >> 1; s > 0; s >>= 1) {
        if (t < s) {
            float a = red[t], b = red[t + s];
            red[t] = (op == 0) ? fminf(a, b) : (op == 1) ? fmaxf(a, b) : (a + b);
        }
        __syncthreads();
    }
    float r = red[0];
    __syncthreads();
    return r;
}

#if USE_TDM
// 1D TDM load: tile of `tile_elems` f32 starting at byte address `gaddr`
// (element offset within a tensor of `total_elems` f32) into LDS at `lds_off`.
// D# layout per CDNA5 ISA ch.8 (group0 128b, group1 256b; groups 2/3 zero => <=2D).
__device__ __forceinline__ void tdm_load_f32_1d(unsigned lds_off,
                                                unsigned long long gaddr,
                                                int total_elems, int tile_elems) {
    u32x4 g0;
    g0.x = 1u;                                               // count=1, user desc
    g0.y = lds_off;                                          // LDS byte address
    g0.z = (unsigned)(gaddr & 0xffffffffull);                // global_addr[31:0]
    g0.w = (unsigned)((gaddr >> 32) & 0x1ffffffull)          // global_addr[56:32]
         | (2u << 30);                                       // type=2 (image)
    i32x8 g1;
    g1[0] = (int)(2u << 16);                                 // data_size=2 (4 bytes)
    g1[1] = (int)(((unsigned)total_elems & 0xffffu) << 16);  // tensor_dim0[15:0]
    g1[2] = (int)((((unsigned)total_elems >> 16) & 0xffffu)  // tensor_dim0[31:16]
         | (1u << 16));                                      // tensor_dim1[15:0]=1
    g1[3] = (int)(((unsigned)tile_elems & 0xffffu) << 16);   // dim1 hi=0 | tile_dim0
    g1[4] = 1;                                               // tile_dim1=1, tile_dim2=0
    g1[5] = total_elems;                                     // tensor_dim0_stride[31:0]
    g1[6] = 0;                                               // stride hi | dim1_stride lo
    g1[7] = 0;
    i32x4 gz = {0, 0, 0, 0};
#if __clang_major__ >= 23
    i32x8 gz8 = {0, 0, 0, 0, 0, 0, 0, 0};
    __builtin_amdgcn_tensor_load_to_lds(g0, g1, gz, gz, gz8, 0);
#else
    __builtin_amdgcn_tensor_load_to_lds(g0, g1, gz, gz, 0);
#endif
}
#endif

// ---------------- k1: per-block masked/global min-max over full images ----------------
__global__ __launch_bounds__(256) void k1_minmax(const float* __restrict__ warped,
                                                 const float* __restrict__ fixedI,
                                                 const float* __restrict__ disp,
                                                 float* __restrict__ part,
                                                 int*   __restrict__ pany,
                                                 int n) {
    float mnFm =  INFINITY, mxFm = -INFINITY, mnWm =  INFINITY, mxWm = -INFINITY;
    float mnFa =  INFINITY, mxFa = -INFINITY, mnWa =  INFINITY, mxWa = -INFINITY;
    int any = 0;
    for (int i = blockIdx.x * blockDim.x + threadIdx.x; i < n; i += gridDim.x * blockDim.x) {
        float f = fixedI[i];
        float w = warped[i];
        size_t di = 3u * (size_t)i;
        float d0 = disp[di], d1 = disp[di + 1], d2 = disp[di + 2];
        bool valid = (fabsf(d0) <= 1.0f) && (fabsf(d1) <= 1.0f) && (fabsf(d2) <= 1.0f);
        mnFa = fminf(mnFa, f); mxFa = fmaxf(mxFa, f);
        mnWa = fminf(mnWa, w); mxWa = fmaxf(mxWa, w);
        if (valid) {
            any = 1;
            mnFm = fminf(mnFm, f); mxFm = fmaxf(mxFm, f);
            mnWm = fminf(mnWm, w); mxWm = fmaxf(mxWm, w);
        }
    }
    __shared__ float red[256];
    float r0 = blockReduceOp(mnFm, 0, red, 256);
    float r1 = blockReduceOp(mxFm, 1, red, 256);
    float r2 = blockReduceOp(mnWm, 0, red, 256);
    float r3 = blockReduceOp(mxWm, 1, red, 256);
    float r4 = blockReduceOp(mnFa, 0, red, 256);
    float r5 = blockReduceOp(mxFa, 1, red, 256);
    float r6 = blockReduceOp(mnWa, 0, red, 256);
    float r7 = blockReduceOp(mxWa, 1, red, 256);
    float ra = blockReduceOp((float)any, 1, red, 256);
    if (threadIdx.x == 0) {
        float* p = part + (size_t)blockIdx.x * 8;
        p[0] = r0; p[1] = r1; p[2] = r2; p[3] = r3;
        p[4] = r4; p[5] = r5; p[6] = r6; p[7] = r7;
        pany[blockIdx.x] = (ra > 0.0f) ? 1 : 0;
    }
}

// ---------------- k2: final reduce -> bin edges; zero accumulators ----------------
__global__ __launch_bounds__(256) void k2_bins(float* __restrict__ ws, int nb) {
    const float* part = ws + WS_PART;
    const int*   pany = (const int*)ws + WS_PANY;
    int t = threadIdx.x;
    float mnFm =  INFINITY, mxFm = -INFINITY, mnWm =  INFINITY, mxWm = -INFINITY;
    float mnFa =  INFINITY, mxFa = -INFINITY, mnWa =  INFINITY, mxWa = -INFINITY;
    int any = 0;
    for (int i = t; i < nb; i += 256) {
        const float* p = part + (size_t)i * 8;
        mnFm = fminf(mnFm, p[0]); mxFm = fmaxf(mxFm, p[1]);
        mnWm = fminf(mnWm, p[2]); mxWm = fmaxf(mxWm, p[3]);
        mnFa = fminf(mnFa, p[4]); mxFa = fmaxf(mxFa, p[5]);
        mnWa = fminf(mnWa, p[6]); mxWa = fmaxf(mxWa, p[7]);
        any |= pany[i];
    }
    __shared__ float red[256];
    __shared__ float sc[8];
    __shared__ int   sany;
    float r0 = blockReduceOp(mnFm, 0, red, 256);
    float r1 = blockReduceOp(mxFm, 1, red, 256);
    float r2 = blockReduceOp(mnWm, 0, red, 256);
    float r3 = blockReduceOp(mxWm, 1, red, 256);
    float r4 = blockReduceOp(mnFa, 0, red, 256);
    float r5 = blockReduceOp(mxFa, 1, red, 256);
    float r6 = blockReduceOp(mnWa, 0, red, 256);
    float r7 = blockReduceOp(mxWa, 1, red, 256);
    float ra = blockReduceOp((float)any, 1, red, 256);
    if (t == 0) {
        sc[0] = r0; sc[1] = r1; sc[2] = r2; sc[3] = r3;
        sc[4] = r4; sc[5] = r5; sc[6] = r6; sc[7] = r7;
        sany = (ra > 0.0f) ? 1 : 0;
        for (int j = 0; j < 8; ++j) ws[WS_SCALARS + j] = sc[j];
        ((int*)ws)[WS_ANY] = sany;
        ws[WS_WSUM] = 0.0f;
    }
    __syncthreads();
    // fallback to all-ones mask if nothing is valid
    float bgF  = sany ? sc[0] : sc[4];
    float maxF = sany ? sc[1] : sc[5];
    float bgW  = sany ? sc[2] : sc[6];
    float maxW = sany ? sc[3] : sc[7];
    if (t < BINS) {
        float stepF = (maxF - bgF) / (float)(BINS - 1);
        float stepW = (maxW - bgW) / (float)(BINS - 1);
        ws[WS_BINSF + t] = bgF + stepF * (float)t;
        ws[WS_BINSW + t] = bgW + stepW * (float)t;
    }
    for (int i = t; i < BINS * BINS; i += 256) ws[WS_JOINT + i] = 0.0f;
    if (t < BINS) { ws[WS_MARGF + t] = 0.0f; ws[WS_MARGW + t] = 0.0f; }
}

// ---------------- k3: hashed spatial sampling (gather, interleaved) + weight sum --------
__global__ __launch_bounds__(256) void k3_gather(const float* __restrict__ warped,
                                                 const float* __restrict__ fixedI,
                                                 const float* __restrict__ disp,
                                                 const float* __restrict__ ws_ro,
                                                 float* __restrict__ samp,   // 3m floats
                                                 float* __restrict__ wsum,
                                                 int n, int m) {
    int any = ((const int*)ws_ro)[WS_ANY];
    int i = blockIdx.x * blockDim.x + threadIdx.x;
    float w = 0.0f;
    if (i < m) {
        unsigned idx = hash32((unsigned)i) % (unsigned)n;
        size_t di = 3u * (size_t)idx;
        float d0 = disp[di], d1 = disp[di + 1], d2 = disp[di + 2];
        bool valid = (fabsf(d0) <= 1.0f) && (fabsf(d1) <= 1.0f) && (fabsf(d2) <= 1.0f);
        w = any ? (valid ? 1.0f : 0.0f) : 1.0f;
        size_t si = 3u * (size_t)i;
        samp[si + 0] = fixedI[idx];
        samp[si + 1] = warped[idx];
        samp[si + 2] = w;
    }
    __shared__ float red[256];
    float s = blockReduceOp(w, 2, red, 256);
    if (threadIdx.x == 0) atomicAdd(wsum, s);
}

// ---------------- k4: WMMA joint histogram (64xM @ Mx64) + marginals ----------------
// 512 threads = 16 wave32 waves; wave wv owns output tile (tr=wv>>2, tc=wv&3).
// Per chunk: TDM double-buffers 128 interleaved samples into LDS; all threads build
// the 64x128 f16 P-panels ONCE in WMMA fragment layout; waves feed v_wmma from LDS.
__global__ __launch_bounds__(512) void k4_joint(const float* __restrict__ samp,
                                                const float* __restrict__ bins_f,
                                                const float* __restrict__ bins_w,
                                                float* __restrict__ joint,
                                                float* __restrict__ margF,
                                                float* __restrict__ margW,
                                                int m, int nch) {
    __shared__ float    sbuf[2][3 * CHUNK];   // staged samples (double buffered)
    __shared__ _Float16 Ah[4 * 4 * 32 * 16];  // P_f * w : [kk][tr][lane][h]
    __shared__ _Float16 Bh[4 * 4 * 32 * 16];  // P_w     : [kk][tc][lane][h]

    const int t    = threadIdx.x;
    const int lane = t & 31;
    const int wv   = t >> 5;          // 0..15
    const int tr   = wv >> 2;         // output tile row
    const int tc   = wv & 3;          // output tile col
    const int lrow = lane & 15;
    const int hi   = lane >> 4;

    // panel-producer role: thread owns bin row (t>>3) and k-slab (t&7)*16
    const int rowM = t >> 3;          // 0..63
    const int k0   = (t & 7) * 16;    // 0..112
    const float bf = bins_f[rowM];
    const float bw = bins_w[rowM];

    const float invS    = 1.0f / SIGMA_F;                     // exponent is -(d^2)/sigma
    const float invNorm = 1.0f / (sqrtf(TWO_PI_F) * SIGMA_F); // 1/(sqrt(2pi)*sigma)
    const int   td0     = 3 * m;                              // tensor length (f32 elems)

    v8f   acc = {};        // 16x16 f32 C/D tile
    float mF = 0.0f;       // weighted p_f partial sum for bin rowM
    float mW = 0.0f;       // weighted p_w partial sum for bin rowM

    const int ch0 = blockIdx.x;
    int cur = 0;

#if USE_TDM
    const unsigned long long sbase = (unsigned long long)(uintptr_t)samp;
    if (wv == 0 && ch0 < nch) {
        tdm_load_f32_1d((unsigned)(uintptr_t)&sbuf[0][0],
                        sbase + (unsigned long long)ch0 * (3ull * CHUNK * 4ull),
                        td0, 3 * CHUNK);
    }
#endif

    for (int ch = ch0; ch < nch; ch += gridDim.x) {
#if USE_TDM
        const int nxt = ch + (int)gridDim.x;
        if (wv == 0) {
            if (nxt < nch) {
                tdm_load_f32_1d((unsigned)(uintptr_t)&sbuf[cur ^ 1][0],
                                sbase + (unsigned long long)nxt * (3ull * CHUNK * 4ull),
                                td0, 3 * CHUNK);
                __builtin_amdgcn_s_wait_tensorcnt(1);  // chunk `ch` landed
            } else {
                __builtin_amdgcn_s_wait_tensorcnt(0);
            }
        }
#else
        if (t < 3 * CHUNK) {
            int g = ch * (3 * CHUNK) + t;
            sbuf[cur][t] = (g < td0) ? samp[g] : 0.0f;
            int nx = g + (int)gridDim.x * (3 * CHUNK);
            if (nx < td0) __builtin_prefetch(&samp[nx], 0, 0);
        }
#endif
        __syncthreads();   // samples ready; previous WMMA reads of Ah/Bh done

        // ---- build P panels once (each element exactly one exp) ----
        const float* S = &sbuf[cur][0];
        #pragma unroll
        for (int r8 = 0; r8 < 2; ++r8) {
            const int ks   = k0 + r8 * 8;        // multiple of 8
            const int kk   = ks >> 5;            // 32-wide K step
            const int kloc = ks & 31;            // 0,8,16,24
            v8h av, bv;
            #pragma unroll
            for (int j = 0; j < 8; ++j) {
                const int k = ks + j;
                const float xf = S[3 * k + 0];
                const float xw = S[3 * k + 1];
                const float wk = S[3 * k + 2];
                float da = xf - bf;
                float pa = __expf(-da * da * invS) * invNorm * wk;  // weighted p_f
                float db = xw - bw;
                float pb = __expf(-db * db * invS) * invNorm;       // unweighted p_w
                av[j] = (_Float16)pa;
                bv[j] = (_Float16)pb;
                mF += pa;
                mW += pb * wk;
            }
            // A: k = h + (h&8) + 8*(lane>=16)  => inverse of kloc
            const int bA   = kloc >> 3;          // 0..3
            const int hi2A = bA & 1;
            const int h0A  = 8 * (bA >> 1);
            const int idxA = (((kk * 4 + (rowM >> 4)) * 32) + (hi2A * 16 + (rowM & 15))) * 16 + h0A;
            *(v8h*)&Ah[idxA] = av;
            // B: k = h + (h&8) + 8*(lane>=16)  => same as A
            const int hi2B = (kloc >> 3) & 1;
            const int h0B  = 8 * (kloc >> 4);    // 0 or 8
            const int idxB = (((kk * 4 + (rowM >> 4)) * 32) + (hi2B * 16 + (rowM & 15))) * 16 + h0B;
            *(v8h*)&Bh[idxB] = bv;
        }
        __syncthreads();   // panels ready

        // ---- WMMA: each wave accumulates its 16x16 tile over K=128 ----
        #pragma unroll
        for (int kk = 0; kk < CHUNK / 32; ++kk) {
            const v16h a = *(const v16h*)&Ah[((kk * 4 + tr) * 32 + lane) * 16];
            const v16h b = *(const v16h*)&Bh[((kk * 4 + tc) * 32 + lane) * 16];
            acc = __builtin_amdgcn_wmma_f32_16x16x32_f16(
                false, a, false, b, (short)0, acc, false, false);
        }
        cur ^= 1;
    }

    // marginals: each (row, k) element was produced exactly once
    atomicAdd(&margF[rowM], mF);
    atomicAdd(&margW[rowM], mW);

    // C/D layout: VGPR i -> (M = i + 8*hi, N = lane&15) within the 16x16 tile
    #pragma unroll
    for (int i = 0; i < 8; ++i) {
        int r = tr * 16 + i + hi * 8;
        int c = tc * 16 + lrow;
        atomicAdd(&joint[r * BINS + c], acc[i]);
    }
}

// ---------------- k5: entropies -> MI scalar ----------------
__global__ __launch_bounds__(64) void k5_final(const float* __restrict__ ws,
                                               float* __restrict__ out) {
    __shared__ float red[64];
    const int t = threadIdx.x;
    const float wsum = ws[WS_WSUM];

    float pf = ws[WS_MARGF + t] / wsum;
    float sF = blockReduceOp(pf, 2, red, 64);
    float pnf = pf / (sF + EPS_F);
    float entF = blockReduceOp(-pnf * log2f(pnf + EPS_F), 2, red, 64);

    float pw = ws[WS_MARGW + t] / wsum;
    float sW = blockReduceOp(pw, 2, red, 64);
    float pnw = pw / (sW + EPS_F);
    float entW = blockReduceOp(-pnw * log2f(pnw + EPS_F), 2, red, 64);

    const float jscale = 1.0f / (TWO_PI_F * SIGMA_F * SIGMA_F);  // 1/(2*pi*sigma^2)
    float js = 0.0f;
    for (int i = t; i < BINS * BINS; i += 64) js += ws[WS_JOINT + i] * jscale;
    float jt = blockReduceOp(js, 2, red, 64);
    float ej = 0.0f;
    for (int i = t; i < BINS * BINS; i += 64) {
        float p = ws[WS_JOINT + i] * jscale / (jt + EPS_F);
        ej += -p * log2f(p + EPS_F);
    }
    float entJ = blockReduceOp(ej, 2, red, 64);

    if (t == 0) out[0] = -(entF + entW - entJ);
}

// ---------------- launch ----------------
extern "C" void kernel_launch(void* const* d_in, const int* in_sizes, int n_in,
                              void* d_out, int out_size, void* d_ws, size_t ws_size,
                              hipStream_t stream) {
    const float* warped = (const float*)d_in[0];
    const float* moving = (const float*)d_in[1]; (void)moving;  // unused by forward
    const float* fixedI = (const float*)d_in[2];
    const float* disp   = (const float*)d_in[3];
    (void)n_in; (void)ws_size; (void)out_size;

    const int n = in_sizes[0];          // 8,028,160
    const int m = n / 10;               // int(0.1*n) = 802,816 (multiple of CHUNK)

    float* ws   = (float*)d_ws;
    float* part = ws + WS_PART;
    int*   pany = (int*)d_ws + WS_PANY;
    float* samp = ws + WS_SAMPLES;      // 3m floats, interleaved

    const int NB = 1024;
    k1_minmax<<<NB, 256, 0, stream>>>(warped, fixedI, disp, part, pany, n);
    k2_bins<<<1, 256, 0, stream>>>(ws, NB);
    k3_gather<<<(m + 255) / 256, 256, 0, stream>>>(warped, fixedI, disp,
                                                   (const float*)ws, samp,
                                                   ws + WS_WSUM, n, m);
    const int nch = (m + CHUNK - 1) / CHUNK;   // 6272 chunks
    k4_joint<<<256, 512, 0, stream>>>(samp,
                                      ws + WS_BINSF, ws + WS_BINSW,
                                      ws + WS_JOINT, ws + WS_MARGF, ws + WS_MARGW,
                                      m, nch);
    k5_final<<<1, 64, 0, stream>>>(ws, (float*)d_out);
}
